// Block_876173328518
// MI455X (gfx1250) — matrix operand order repaired and checked by reading, not compiled, not run on hardware
//
#include <hip/hip_runtime.h>

typedef __bf16 bf16;
typedef __attribute__((ext_vector_type(16))) __bf16 v16bf;
typedef __attribute__((ext_vector_type(8)))  __bf16 bf16x8;
typedef __attribute__((ext_vector_type(8)))  float  v8f;
typedef unsigned int u32x4 __attribute__((ext_vector_type(4)));
typedef int i32x4 __attribute__((ext_vector_type(4)));
typedef int i32x8 __attribute__((ext_vector_type(8)));

#define T_SEQ 2048
#define C_EMB 2048
#define N_HEADS 16
#define HDIM 128
#define CQKV 6144
#define HID 5632

#define AS1 __attribute__((address_space(1)))
#define AS3 __attribute__((address_space(3)))

// ---------------------------------------------------------------------------
// gfx1250 async copy: global -> LDS (ASYNCcnt).
// Builtin signature (from hipcc diagnostic): (int4 AS1*, int4 AS3*, imm, imm)
// ---------------------------------------------------------------------------
__device__ __forceinline__ void async_ld_b128(void* lds, const void* g) {
#if __has_builtin(__builtin_amdgcn_global_load_async_to_lds_b128)
  __builtin_amdgcn_global_load_async_to_lds_b128(
      (AS1 i32x4*)(unsigned long long)g,
      (AS3 i32x4*)(unsigned)(unsigned long long)lds, 0, 0);
#else
  asm volatile("global_load_async_to_lds_b128 %0, %1, off"
               :: "v"((unsigned)(unsigned long long)lds), "v"(g)
               : "memory");
#endif
}

__device__ __forceinline__ void wait_async0() {
#if __has_builtin(__builtin_amdgcn_s_wait_asynccnt)
  __builtin_amdgcn_s_wait_asynccnt(0);
#else
  asm volatile("s_wait_asynccnt 0" ::: "memory");
#endif
}

// ---------------------------------------------------------------------------
// Tensor Data Mover: one 2-D tile (bf16) global -> LDS, tracked by TENSORcnt.
// D# packing per ISA 8.3/8.4.
// ---------------------------------------------------------------------------
#if __has_builtin(__builtin_amdgcn_tensor_load_to_lds)
#define HAVE_TDM 1
__device__ __forceinline__ void tdm_load_tile_2d(const void* gaddr, unsigned lds_off,
                                                 unsigned tile_d0, unsigned tile_d1,
                                                 unsigned tens_d0, unsigned tens_d1,
                                                 unsigned long long stride0) {
  unsigned long long ga = (unsigned long long)gaddr;
  u32x4 g0 = {};
  g0[0] = 1u;                                                  // count=1 (valid)
  g0[1] = lds_off;                                             // lds_addr
  g0[2] = (unsigned)ga;                                        // global_addr[31:0]
  g0[3] = (unsigned)((ga >> 32) & 0x01ffffffu) | (2u << 30);   // ga[56:32] | type=2
  i32x8 g1 = {};
  g1[0] = (int)(1u << 16);                                     // data_size = 2 bytes
  g1[1] = (int)((tens_d0 & 0xffffu) << 16);                    // tensor_dim0[15:0]
  g1[2] = (int)((tens_d0 >> 16) | ((tens_d1 & 0xffffu) << 16));// td0[31:16]|td1[15:0]
  g1[3] = (int)((tens_d1 >> 16) | ((tile_d0 & 0xffffu) << 16));// td1[31:16]|tile_d0
  g1[4] = (int)(tile_d1 & 0xffffu);                            // tile_dim1, tile_dim2=0
  g1[5] = (int)(unsigned)stride0;                              // dim0_stride[31:0]
  g1[6] = (int)((unsigned)(stride0 >> 32) & 0xffffu);          // dim0_stride[47:32]
  g1[7] = 0;
  i32x4 z4 = {};
#if __clang_major__ >= 23
  i32x8 z8 = {};
  __builtin_amdgcn_tensor_load_to_lds(g0, g1, z4, z4, z8, 0);
#else
  __builtin_amdgcn_tensor_load_to_lds(g0, g1, z4, z4, 0);
#endif
}
#endif

// ---------------------------------------------------------------------------
// f32 -> bf16 cast (weights)
// ---------------------------------------------------------------------------
__global__ __launch_bounds__(256) void k_cast_bf16(const float* __restrict__ src,
                                                   bf16* __restrict__ dst, int n) {
  int i = blockIdx.x * 256 + threadIdx.x;
  if (i < n) dst[i] = (bf16)src[i];
}

// ---------------------------------------------------------------------------
// RMSNorm: one block (8 waves) per row; emits bf16
// ---------------------------------------------------------------------------
__global__ __launch_bounds__(256) void k_rmsnorm(const float* __restrict__ x,
                                                 const float* __restrict__ g,
                                                 bf16* __restrict__ out, int C) {
  int row = blockIdx.x;
  int tid = threadIdx.x;
  __shared__ float red[8];
  const float* xr = x + (size_t)row * C;
  float ss = 0.f;
  for (int c = tid; c < C; c += 256) { float v = xr[c]; ss += v * v; }
#pragma unroll
  for (int m = 16; m >= 1; m >>= 1) ss += __shfl_xor(ss, m, 32);
  if ((tid & 31) == 0) red[tid >> 5] = ss;
  __syncthreads();
  float tot = 0.f;
#pragma unroll
  for (int w = 0; w < 8; ++w) tot += red[w];
  float rstd = rsqrtf(tot / (float)C + 1e-5f);
  bf16* orow = out + (size_t)row * C;
  for (int c = tid; c < C; c += 256) orow[c] = (bf16)(xr[c] * rstd * g[c]);
}

// ---------------------------------------------------------------------------
// bf16 WMMA GEMM: out[M,N] = A[M,K] @ B[K,N]  (row-major)
// Tile 128x128x32, 256 thr = 8 waves (2x4), 4x2 WMMA tiles per wave.
// Double-buffered pipeline: A tile i+1 via TDM + B tile i+1 via global b128
// loads overlap the WMMAs of tile i. B stored transposed in LDS so both
// operand fragments are contiguous ds_load_b128 reads. One barrier per tile.
// epi: 0 = f32 out, 1 = f32 out + residual, 2 = bf16 out
// ---------------------------------------------------------------------------
__global__ __launch_bounds__(256) void k_gemm(const bf16* __restrict__ A,
                                              const bf16* __restrict__ B,
                                              int M, int N, int K,
                                              const float* __restrict__ res,
                                              float* __restrict__ outF,
                                              bf16* __restrict__ outH, int epi) {
  __shared__ __align__(16) bf16 sA [2][128 * 32];   // [M][K]
  __shared__ __align__(16) bf16 sBT[2][128 * 32];   // [N][K] (transposed)
  const int bx = blockIdx.x;           // N tile
  const int by = blockIdx.y;           // M tile
  const int tid = threadIdx.x;
  const int lane = tid & 31;
  const int wave = tid >> 5;
  const int wm = wave >> 2;            // 0..1
  const int wn = wave & 3;             // 0..3
  const int m16 = lane & 15;
  const int kb  = (lane >> 4) * 8;     // A-frag K sub-block select
  const int kh  = (lane >> 4) * 16;    // K half offset; unused (B-frag uses kb)
  const int brow = tid >> 4;           // B stage: k row 0..15 (+16 for 2nd chunk)
  const int bcol = (tid & 15) * 8;     // B stage: n col base

  v8f acc[4][2] = {};
  const int nt = K >> 5;

  // ---- tile-load helpers ----
  auto issueA = [&](int i, int buf) {
#ifdef HAVE_TDM
    if (wave == 0)
      tdm_load_tile_2d(&A[(size_t)(by * 128) * K + i * 32],
                       (unsigned)(unsigned long long)(void*)&sA[buf][0],
                       32u, 128u, (unsigned)K, (unsigned)M, (unsigned long long)K);
#else
#pragma unroll
    for (int u = 0; u < 2; ++u) {
      int idx = tid + u * 256;
      int r = idx >> 2;
      int c = (idx & 3) * 8;
      async_ld_b128(&sA[buf][r * 32 + c], &A[(size_t)(by * 128 + r) * K + i * 32 + c]);
    }
#endif
  };
  auto loadB = [&](int i, bf16x8& a0, bf16x8& a1) {
    a0 = *(const bf16x8*)&B[(size_t)(i * 32 + brow) * N + bx * 128 + bcol];
    a1 = *(const bf16x8*)&B[(size_t)(i * 32 + brow + 16) * N + bx * 128 + bcol];
  };
  auto storeBT = [&](int buf, bf16x8 a0, bf16x8 a1) {
#pragma unroll
    for (int t = 0; t < 8; ++t) sBT[buf][(bcol + t) * 32 + brow] = a0[t];
#pragma unroll
    for (int t = 0; t < 8; ++t) sBT[buf][(bcol + t) * 32 + brow + 16] = a1[t];
  };

  // ---- prologue: tile 0 ----
  bf16x8 r0, r1;
  issueA(0, 0);
  loadB(0, r0, r1);
  storeBT(0, r0, r1);

  for (int i = 0; i < nt; ++i) {
    const int bufc = i & 1, bufn = bufc ^ 1;
    const bool more = (i + 1 < nt);

    // wait for this tile's DMA, align all waves (barrier also orders BT stores)
#ifdef HAVE_TDM
    if (wave == 0) __builtin_amdgcn_s_wait_tensorcnt(0);
#else
    wait_async0();
#endif
    __syncthreads();

    // kick off tile i+1 into the other buffer (overlaps WMMAs below)
    if (more) {
      issueA(i + 1, bufn);
      loadB(i + 1, r0, r1);
    }
    if (i + 2 < nt) {
      __builtin_prefetch(&A[(size_t)(by * 128 + (tid >> 1)) * K + (i + 2) * 32 + (tid & 1) * 16], 0, 3);
      __builtin_prefetch(&B[(size_t)((i + 2) * 32 + brow) * N + bx * 128 + bcol], 0, 3);
    }

    // ---- fragments (ISA 7.12.2 bf16 layouts), all contiguous LDS reads ----
    v16bf af[4], bfr[2];
#pragma unroll
    for (int mt = 0; mt < 4; ++mt) {
      int row = wm * 64 + mt * 16 + m16;
#pragma unroll
      for (int j = 0; j < 16; ++j) {
        int v = j >> 1;
        int k = ((v < 4) ? 2 * v : 16 + 2 * (v - 4)) + kb + (j & 1);
        af[mt][j] = sA[bufc][row * 32 + k];
      }
    }
#pragma unroll
    for (int ntl = 0; ntl < 2; ++ntl) {
      int col = wn * 32 + ntl * 16 + m16;
      const bf16* bb = &sBT[bufc][col * 32 + kb];
#pragma unroll
      for (int j = 0; j < 16; ++j) bfr[ntl][j] = bb[(j < 8) ? j : j + 8];
    }

#pragma unroll
    for (int mt = 0; mt < 4; ++mt)
#pragma unroll
      for (int ntl = 0; ntl < 2; ++ntl)
        acc[mt][ntl] = __builtin_amdgcn_wmma_f32_16x16x32_bf16(
            false, af[mt], false, bfr[ntl], (short)0, acc[mt][ntl], false, false);

    if (more) storeBT(bufn, r0, r1);
  }

  // ---- epilogue: D layout = VGPR i, lanes0-15 row i, lanes16-31 row i+8 ----
  const int colh = lane & 15;
  const int rofs = (lane >> 4) * 8;
#pragma unroll
  for (int mt = 0; mt < 4; ++mt)
#pragma unroll
    for (int ntl = 0; ntl < 2; ++ntl)
#pragma unroll
      for (int i = 0; i < 8; ++i) {
        int row = by * 128 + wm * 64 + mt * 16 + rofs + i;
        int col = bx * 128 + wn * 32 + ntl * 16 + colh;
        size_t o = (size_t)row * N + col;
        float v = acc[mt][ntl][i];
        if (epi == 1) v += res[o];
        if (epi == 2) outH[o] = (bf16)v;
        else          outF[o] = v;
      }
}

// ---------------------------------------------------------------------------
// Flash attention with ALiBi + causal mask.
// 4 waves per block, each owning one 16-query tile; the 32-key K/V staging is
// shared by all 4 waves (4x HBM reuse). K staged via async b128 copies; V
// staged transposed so P@V B-fragments are contiguous ds_load_b128 reads.
// ---------------------------------------------------------------------------
__global__ __launch_bounds__(128) void k_attn(const bf16* __restrict__ qkv,
                                              const float* __restrict__ alibi,
                                              bf16* __restrict__ yout) {
  __shared__ __align__(16) bf16 sK [32 * 128];    // [key][d]
  __shared__ __align__(16) bf16 sVT[128 * 32];    // [d][key] (transposed)
  __shared__ __align__(16) bf16 sP [4][16 * 32];  // per-wave P tile
  const int tid  = threadIdx.x;
  const int lane = tid & 31;
  const int wave = tid >> 5;
  const int h  = blockIdx.y;
  const int qt = blockIdx.x * 4 + wave;
  const int m16 = lane & 15;
  const int kb  = (lane >> 4) * 8;
  const int kh16 = (lane >> 4) * 16;
  const int colh = lane & 15;
  const int rofs = (lane >> 4) * 8;
  const float scale = 0.08838834764831845f;  // 1/sqrt(128)

  // ---- preload Q fragments: 4 chunks of K=32 over hd=128 ----
  v16bf qf[4];
  {
    const int qrow = qt * 16 + m16;
    const bf16* qp = qkv + (size_t)qrow * CQKV + h * HDIM;
#pragma unroll
    for (int kc = 0; kc < 4; ++kc)
#pragma unroll
      for (int j = 0; j < 16; ++j) {
        int v = j >> 1;
        int k = ((v < 4) ? 2 * v : 16 + 2 * (v - 4)) + kb + (j & 1);
        qf[kc][j] = qp[kc * 32 + k];
      }
  }

  v8f yacc[8] = {};
  float m[8], l[8];
#pragma unroll
  for (int i = 0; i < 8; ++i) { m[i] = -1e30f; l[i] = 0.f; }

  const int nchunks_w = ((qt * 16 + 15) >> 5) + 1;        // this wave's causal range
  const int nchunks_b = ((blockIdx.x * 64 + 63) >> 5) + 1;// block-wide range

  for (int c = 0; c < nchunks_b; ++c) {
    const int k0 = c * 32;

    // ---- cooperative staging: K async, V transposed from registers ----
#pragma unroll
    for (int i = 0; i < 4; ++i) {
      int idx = tid + i * 128;           // 0..511
      int r = idx >> 4;                  // key row 0..31
      int cc = (idx & 15) * 8;           // d col base
      async_ld_b128(&sK[r * 128 + cc],
                    qkv + (size_t)(k0 + r) * CQKV + C_EMB + h * HDIM + cc);
      bf16x8 v = *(const bf16x8*)(qkv + (size_t)(k0 + r) * CQKV + 2 * C_EMB + h * HDIM + cc);
#pragma unroll
      for (int t = 0; t < 8; ++t) sVT[(cc + t) * 32 + r] = v[t];
    }
    wait_async0();
    __syncthreads();

    if (c < nchunks_w) {
      // ---- S = Q @ K^T for 2 x 16 keys ----
      v8f sc[2] = {};
#pragma unroll
      for (int hhalf = 0; hhalf < 2; ++hhalf) {
        const bf16* kbase = &sK[(hhalf * 16 + m16) * 128];
#pragma unroll
        for (int kc = 0; kc < 4; ++kc) {
          v16bf kf;
#pragma unroll
          for (int j = 0; j < 16; ++j) kf[j] = kbase[kc * 32 + kb + ((j < 8) ? j : j + 8)];
          sc[hhalf] = __builtin_amdgcn_wmma_f32_16x16x32_bf16(
              false, qf[kc], false, kf, (short)0, sc[hhalf], false, false);
        }
      }

      // ---- scale + alibi + causal mask + online softmax ----
#pragma unroll
      for (int i = 0; i < 8; ++i) {
        const int grow = qt * 16 + rofs + i;
        const float* arow = alibi + (size_t)h * T_SEQ * T_SEQ + (size_t)grow * T_SEQ;
#pragma unroll
        for (int hhalf = 0; hhalf < 2; ++hhalf) {
          int gcol = k0 + hhalf * 16 + colh;
          float v = sc[hhalf][i] * scale + arow[gcol];
          if (gcol > grow) v = -1e30f;
          sc[hhalf][i] = v;
        }
        float rm = fmaxf(sc[0][i], sc[1][i]);
#pragma unroll
        for (int ms = 1; ms <= 8; ms <<= 1) rm = fmaxf(rm, __shfl_xor(rm, ms, 16));
        float mnew = fmaxf(m[i], rm);
        float al = __expf(m[i] - mnew);
        float p0 = __expf(sc[0][i] - mnew);
        float p1 = __expf(sc[1][i] - mnew);
        sc[0][i] = p0; sc[1][i] = p1;
        float rs = p0 + p1;
#pragma unroll
        for (int ms = 1; ms <= 8; ms <<= 1) rs += __shfl_xor(rs, ms, 16);
        l[i] = l[i] * al + rs;
        m[i] = mnew;
#pragma unroll
        for (int ci = 0; ci < 8; ++ci) yacc[ci][i] *= al;
      }

      // ---- relayout P through per-wave LDS (same-wave DS ops are in-order) ----
#pragma unroll
      for (int i = 0; i < 8; ++i)
#pragma unroll
        for (int hhalf = 0; hhalf < 2; ++hhalf)
          sP[wave][(rofs + i) * 32 + hhalf * 16 + colh] = (bf16)sc[hhalf][i];
      v16bf pf;
#pragma unroll
      for (int j = 0; j < 16; ++j) {
        int v = j >> 1;
        int k = ((v < 4) ? 2 * v : 16 + 2 * (v - 4)) + kb + (j & 1);
        pf[j] = sP[wave][m16 * 32 + k];
      }

      // ---- y += P @ V  (8 column chunks of 16 over hd=128) ----
#pragma unroll
      for (int ci = 0; ci < 8; ++ci) {
        v16bf vf;
        const bf16* vb = &sVT[(ci * 16 + m16) * 32 + kb];
#pragma unroll
        for (int j = 0; j < 16; ++j) vf[j] = vb[(j < 8) ? j : j + 8];
        yacc[ci] = __builtin_amdgcn_wmma_f32_16x16x32_bf16(
            false, pf, false, vf, (short)0, yacc[ci], false, false);
      }
    }
    __syncthreads();
  }

  // ---- normalize + store bf16 ----
#pragma unroll
  for (int ci = 0; ci < 8; ++ci)
#pragma unroll
    for (int i = 0; i < 8; ++i) {
      int grow = qt * 16 + rofs + i;
      yout[(size_t)grow * C_EMB + h * HDIM + ci * 16 + colh] =
          (bf16)(yacc[ci][i] / l[i]);
    }
}

// ---------------------------------------------------------------------------
// SiLU(h1) * h2 -> bf16
// ---------------------------------------------------------------------------
__global__ __launch_bounds__(256) void k_silu_mul(const float* __restrict__ h1,
                                                  const float* __restrict__ h2,
                                                  bf16* __restrict__ out, int n) {
  int i = blockIdx.x * 256 + threadIdx.x;
  if (i < n) {
    float a = h1[i];
    float s = a / (1.f + __expf(-a));
    out[i] = (bf16)(s * h2[i]);
  }
}

// ---------------------------------------------------------------------------
extern "C" void kernel_launch(void* const* d_in, const int* in_sizes, int n_in,
                              void* d_out, int out_size, void* d_ws, size_t ws_size,
                              hipStream_t stream) {
  const float* x      = (const float*)d_in[0];
  const float* alibi  = (const float*)d_in[1];
  const float* w_attn = (const float*)d_in[2];
  const float* w_proj = (const float*)d_in[3];
  const float* w_fc1  = (const float*)d_in[4];
  const float* w_fc2  = (const float*)d_in[5];
  const float* w_mlp  = (const float*)d_in[6];
  const float* rms1   = (const float*)d_in[7];
  const float* rms2   = (const float*)d_in[8];
  float* out = (float*)d_out;

  const int T = T_SEQ, C = C_EMB;

  // bump allocator over d_ws
  char* p = (char*)d_ws;
  auto alloc = [&](size_t bytes) -> void* {
    void* r = (void*)p;
    p += (bytes + 255) & ~(size_t)255;
    return r;
  };
  bf16* wattn_h = (bf16*)alloc((size_t)C * CQKV * 2);
  bf16* wproj_h = (bf16*)alloc((size_t)C * C * 2);
  bf16* wfc1_h  = (bf16*)alloc((size_t)C * HID * 2);
  bf16* wfc2_h  = (bf16*)alloc((size_t)C * HID * 2);
  bf16* wmlp_h  = (bf16*)alloc((size_t)HID * C * 2);
  bf16* xn      = (bf16*)alloc((size_t)T * C * 2);
  bf16* qkv_h   = (bf16*)alloc((size_t)T * CQKV * 2);
  bf16* y_h     = (bf16*)alloc((size_t)T * C * 2);
  float* ares   = (float*)alloc((size_t)T * C * 4);
  float* h1     = (float*)alloc((size_t)T * HID * 4);
  float* h2     = (float*)alloc((size_t)T * HID * 4);
  bf16* hs      = (bf16*)alloc((size_t)T * HID * 2);

  auto cgrid = [](int n) { return (unsigned)((n + 255) / 256); };

  // weights -> bf16
  k_cast_bf16<<<cgrid(C * CQKV), 256, 0, stream>>>(w_attn, wattn_h, C * CQKV);
  k_cast_bf16<<<cgrid(C * C),    256, 0, stream>>>(w_proj, wproj_h, C * C);
  k_cast_bf16<<<cgrid(C * HID),  256, 0, stream>>>(w_fc1,  wfc1_h,  C * HID);
  k_cast_bf16<<<cgrid(C * HID),  256, 0, stream>>>(w_fc2,  wfc2_h,  C * HID);
  k_cast_bf16<<<cgrid(HID * C),  256, 0, stream>>>(w_mlp,  wmlp_h,  HID * C);

  // x -> rmsnorm1 -> bf16
  k_rmsnorm<<<T, 256, 0, stream>>>(x, rms1, xn, C);

  // qkv = xn @ w_attn  (bf16 out)
  k_gemm<<<dim3(CQKV / 128, T / 128), 256, 0, stream>>>(
      xn, wattn_h, T, CQKV, C, nullptr, nullptr, qkv_h, 2);

  // flash attention with alibi (4 query tiles per block share K/V staging)
  k_attn<<<dim3(T / 64, N_HEADS), 128, 0, stream>>>(qkv_h, alibi, y_h);

  // ares = x + y @ w_proj  (f32 + residual)
  k_gemm<<<dim3(C / 128, T / 128), 256, 0, stream>>>(
      y_h, wproj_h, T, C, C, x, ares, nullptr, 1);

  // rmsnorm2
  k_rmsnorm<<<T, 256, 0, stream>>>(ares, rms2, xn, C);

  // h1 = xn @ w_fc1 ; h2 = xn @ w_fc2
  k_gemm<<<dim3(HID / 128, T / 128), 256, 0, stream>>>(
      xn, wfc1_h, T, HID, C, nullptr, h1, nullptr, 0);
  k_gemm<<<dim3(HID / 128, T / 128), 256, 0, stream>>>(
      xn, wfc2_h, T, HID, C, nullptr, h2, nullptr, 0);

  // hs = silu(h1) * h2 (bf16)
  k_silu_mul<<<cgrid(T * HID), 256, 0, stream>>>(h1, h2, hs, T * HID);

  // out = ares + hs @ w_mlp
  k_gemm<<<dim3(C / 128, T / 128), 256, 0, stream>>>(
      hs, wmlp_h, T, C, HID, ares, out, nullptr, 1);
}
